// RecurrentPPOActorCritic_45122926412228
// MI455X (gfx1250) — matrix operand order repaired and checked by reading, not compiled, not run on hardware
//
#include <hip/hip_runtime.h>
#include <hip/hip_bf16.h>

// ---------------------------------------------------------------------------
// Types & WMMA helpers (CDNA5 / gfx1250, wave32)
// ---------------------------------------------------------------------------
typedef __attribute__((ext_vector_type(16))) __bf16 v16bf;
typedef __attribute__((ext_vector_type(8)))  float  v8f;

union Frag16 { v16bf v; unsigned int u[8]; };
static_assert(sizeof(Frag16) == 32, "frag size");

__device__ __forceinline__ unsigned short f2bf(float x) {
  union { float f; unsigned int u; } q; q.f = x;
  unsigned int r = q.u + 0x7FFFu + ((q.u >> 16) & 1u);   // round-to-nearest-even
  return (unsigned short)(r >> 16);
}
__device__ __forceinline__ float bf2f(unsigned short h) {
  union { float f; unsigned int u; } q; q.u = ((unsigned int)h) << 16; return q.f;
}

// 16-bit A-matrix 16x32 layout (ISA 7.12.2): lane 0-15 -> M=lane, K pairs:
// v0..3 = K 0..7, v4..7 = K 16..23 ; lanes 16-31 add +8 to K.
__device__ __forceinline__ int a_kofs(int lane, int v) {
  return ((v < 4) ? (2 * v) : (16 + 2 * (v - 4))) + ((lane & 16) ? 8 : 0);
}
// 16-bit B-matrix 32x16 layout: lane%16 = N, same K order as A:
// v0..3 = K 0..7, v4..7 = K 16..23 ; lanes 16-31 add +8 to K.
__device__ __forceinline__ int b_kofs(int lane, int v) {
  return ((v < 4) ? (2 * v) : (16 + 2 * (v - 4))) + ((lane & 16) ? 8 : 0);
}

__device__ __forceinline__ v8f v8f_zero() {
  v8f z;
#pragma unroll
  for (int i = 0; i < 8; ++i) z[i] = 0.0f;
  return z;
}

__device__ __forceinline__ v8f wmma_bf16(const Frag16& a, const Frag16& b, v8f c) {
  return __builtin_amdgcn_wmma_f32_16x16x32_bf16(
      /*neg_a=*/false, a.v, /*neg_b=*/false, b.v,
      /*c_mod=*/(short)0, c, /*reuse_a=*/false, /*reuse_b=*/false);
}

// ---------------------------------------------------------------------------
// Async global->LDS staging (CDNA5 GLOBAL_LOAD_ASYNC_TO_LDS_B128, ASYNCcnt).
// Each of 256 threads moves 16B per iteration; byte count must divide 4096.
// LDS operand = low 32 bits of the generic pointer (aperture: addr[31:0]).
// ---------------------------------------------------------------------------
__device__ __forceinline__ void stage_to_lds_async(unsigned short* sDst,
                                                   const unsigned short* gSrc,
                                                   int bytes, int tid) {
  unsigned int lds0 = (unsigned int)(size_t)sDst;
  const char* g = (const char*)gSrc;
  for (int off = tid * 16; off < bytes; off += 256 * 16) {
    unsigned int l = lds0 + (unsigned int)off;
    const void* ga = (const void*)(g + off);
    asm volatile("global_load_async_to_lds_b128 %0, %1, off"
                 :: "v"(l), "v"(ga)
                 : "memory");
  }
  asm volatile("s_wait_asynccnt 0x0" ::: "memory");
}

// ---------------------------------------------------------------------------
// Problem constants
// ---------------------------------------------------------------------------
#define TT 128
#define BB 16
#define TB 2048          // T*B
#define NUM_ACTIONS 18
#define FEAT 512
#define HID 256

// ---------------------------------------------------------------------------
// Weight preparation: f32 -> bf16 with layout remap
// ---------------------------------------------------------------------------
template<int Cout, int Cin, int KH, int KW>
__global__ void __launch_bounds__(256)
k_prep_conv_w(const float* __restrict__ w, unsigned short* __restrict__ out) {
  // src (O, I, KH, KW)  ->  dst (O, K) with k = (ky*KW + kx)*Cin + c  (NHWC order)
  constexpr int K = KH * KW * Cin;
  int i = blockIdx.x * 256 + threadIdx.x;
  if (i >= Cout * K) return;
  int o  = i / K;        int k = i - o * K;
  int ky = k / (KW*Cin); int r = k - ky * (KW*Cin);
  int kx = r / Cin;      int c = r - kx * Cin;
  out[i] = f2bf(w[(((o * Cin + c) * KH) + ky) * KW + kx]);
}

__global__ void __launch_bounds__(256)
k_prep_fc_w(const float* __restrict__ w, unsigned short* __restrict__ out) {
  // fc_w (512, 3136) with src k = c*49 + hw (NCHW flatten); dst k = hw*64 + c
  int i = blockIdx.x * 256 + threadIdx.x;
  if (i >= FEAT * 3136) return;
  int o = i / 3136; int k = i - o * 3136;
  int hw = k >> 6;  int c = k & 63;
  out[i] = f2bf(w[o * 3136 + c * 49 + hw]);
}

__global__ void __launch_bounds__(256)
k_prep_plain(const float* __restrict__ w, unsigned short* __restrict__ out, int n) {
  int i = blockIdx.x * 256 + threadIdx.x;
  if (i < n) out[i] = f2bf(w[i]);
}

// ---------------------------------------------------------------------------
// obs: f32 NCHW -> bf16 NHWC, scaled by 1/255
// ---------------------------------------------------------------------------
__global__ void __launch_bounds__(256)
k_obs_to_bf16(const float* __restrict__ obs, unsigned short* __restrict__ out) {
  int i = blockIdx.x * 256 + threadIdx.x;   // exact grid: TB*84*84*4 elements
  int c   = i & 3;
  int pix = i >> 2;
  int w   = pix % 84;
  int t2  = pix / 84;
  int h   = t2 % 84;
  int n   = t2 / 84;
  float v = obs[(((size_t)n * 4 + c) * 84 + h) * 84 + w] * (1.0f / 255.0f);
  out[i] = f2bf(v);
}

// ---------------------------------------------------------------------------
// Implicit-GEMM conv + bias + relu, bf16 in / bf16 out (NHWC), WMMA core.
// One wave computes a 16(M pixels) x Cout tile; 8 waves per block.
// Weights (Cout, K) bf16 staged to LDS via async-to-LDS (no VGPR bounce).
// ---------------------------------------------------------------------------
template<int Cin, int Hin, int Win, int KH, int KW, int S, int Hout, int Wout, int Cout>
__global__ void __launch_bounds__(256)
k_conv_wmma(const unsigned short* __restrict__ X,    // bf16 NHWC input
            const unsigned short* __restrict__ Wbf,  // bf16 (Cout, K)
            const float* __restrict__ bias,
            unsigned short* __restrict__ Y) {        // bf16 NHWC output
  constexpr int K      = KH * KW * Cin;
  constexpr int RowK   = KW * Cin;        // contiguous span per kernel row (mult of 32)
  constexpr int KSTEPS = K / 32;
  constexpr int NT     = Cout / 16;
  static_assert(RowK % 32 == 0, "kernel row must be a multiple of 32 bf16");
  static_assert((Cout * K * 2) % (256 * 16) == 0, "weight bytes must tile 16B x 256");

  __shared__ unsigned short sW[Cout * K];
  const int tid = threadIdx.x;
  stage_to_lds_async(sW, Wbf, Cout * K * 2, tid);
  __syncthreads();

  const int lane = tid & 31;
  const int wv   = tid >> 5;
  const int m0   = (blockIdx.x * 8 + wv) * 16;
  const int mrow = m0 + (lane & 15);
  const int img  = mrow / (Hout * Wout);
  const int rem  = mrow - img * (Hout * Wout);
  const int oy   = rem / Wout;
  const int ox   = rem - oy * Wout;
  const unsigned short* abase =
      X + (((size_t)img * Hin + oy * S) * Win + ox * S) * Cin;

  v8f acc[NT];
#pragma unroll
  for (int t = 0; t < NT; ++t) acc[t] = v8f_zero();

  for (int kk = 0; kk < KSTEPS; ++kk) {
    Frag16 a;
#pragma unroll
    for (int v = 0; v < 8; ++v) {
      int kg = kk * 32 + a_kofs(lane, v);
      int ky = kg / RowK;
      int r  = kg - ky * RowK;
      a.u[v] = *(const unsigned int*)(abase + ky * (Win * Cin) + r);
    }
#pragma unroll
    for (int t = 0; t < NT; ++t) {
      Frag16 b;
      const unsigned short* bp = sW + ((lane & 15) + 16 * t) * K + kk * 32;
#pragma unroll
      for (int v = 0; v < 8; ++v)
        b.u[v] = *(const unsigned int*)(bp + b_kofs(lane, v));
      acc[t] = wmma_bf16(a, b, acc[t]);
    }
  }

  const int nl  = lane & 15;
  const int mhi = (lane & 16) ? 8 : 0;
#pragma unroll
  for (int t = 0; t < NT; ++t) {
    int n = nl + 16 * t;
    float bv = bias[n];
#pragma unroll
    for (int v = 0; v < 8; ++v) {
      int m = m0 + mhi + v;
      float val = acc[t][v] + bv;
      val = val > 0.0f ? val : 0.0f;
      Y[(size_t)m * Cout + n] = f2bf(val);
    }
  }
}

// ---------------------------------------------------------------------------
// Generic WMMA GEMM: out(M,N) = A(M,K)bf16 @ B(N,K)bf16^T + bias[ + bias2]
// RELU_BF: relu + bf16 store; else f32 store. 8 waves/block, 16 M-rows/wave,
// 64-column chunks (4 accumulators) per pass. global_prefetch for streaming.
// ---------------------------------------------------------------------------
template<int K, int NTOT, bool RELU_BF, bool TWO_BIAS>
__global__ void __launch_bounds__(256)
k_gemm_bf16(const unsigned short* __restrict__ A,
            const unsigned short* __restrict__ B,
            const float* __restrict__ bias1,
            const float* __restrict__ bias2,
            unsigned short* __restrict__ outBF,
            float* __restrict__ outF) {
  constexpr int KSTEPS = K / 32;
  constexpr int NCH    = NTOT / 64;
  const int tid  = threadIdx.x;
  const int lane = tid & 31;
  const int wv   = tid >> 5;
  const int m0   = (blockIdx.x * 8 + wv) * 16;
  const unsigned short* arow = A + (size_t)(m0 + (lane & 15)) * K;
  const int nl  = lane & 15;
  const int mhi = (lane & 16) ? 8 : 0;

  for (int nc = 0; nc < NCH; ++nc) {
    v8f acc[4];
#pragma unroll
    for (int t = 0; t < 4; ++t) acc[t] = v8f_zero();
    for (int kk = 0; kk < KSTEPS; ++kk) {
      if (kk + 2 < KSTEPS) {
        __builtin_prefetch(arow + (kk + 2) * 32, 0, 3);         // global_prefetch_b8
        __builtin_prefetch(B + (size_t)(nc * 64 + nl) * K + (kk + 2) * 32, 0, 3);
      }
      Frag16 a;
#pragma unroll
      for (int v = 0; v < 8; ++v)
        a.u[v] = *(const unsigned int*)(arow + kk * 32 + a_kofs(lane, v));
#pragma unroll
      for (int t = 0; t < 4; ++t) {
        Frag16 b;
        const unsigned short* bp = B + (size_t)(nc * 64 + 16 * t + nl) * K + kk * 32;
#pragma unroll
        for (int v = 0; v < 8; ++v)
          b.u[v] = *(const unsigned int*)(bp + b_kofs(lane, v));
        acc[t] = wmma_bf16(a, b, acc[t]);
      }
    }
#pragma unroll
    for (int t = 0; t < 4; ++t) {
      int n = nc * 64 + 16 * t + nl;
      float bv = bias1[n];
      if (TWO_BIAS) bv += bias2[n];
#pragma unroll
      for (int v = 0; v < 8; ++v) {
        int m = m0 + mhi + v;
        float val = acc[t][v] + bv;
        if (RELU_BF) {
          val = val > 0.0f ? val : 0.0f;
          outBF[(size_t)m * NTOT + n] = f2bf(val);
        } else {
          outF[(size_t)m * NTOT + n] = val;
        }
      }
    }
  }
}

// ---------------------------------------------------------------------------
// LSTM scan: single persistent block, 8 waves. Per step:
//   mask state -> gates(16,1024) = h_bf(16,256) @ Whh^T (WMMA) -> LDS
//   -> add precomputed GX[t] -> sigmoid/tanh update -> h to HOUT.
// ---------------------------------------------------------------------------
__global__ void __launch_bounds__(256)
k_lstm(const float* __restrict__ GX,            // (TB, 1024) = x@W_ih^T + b_ih + b_hh
       const unsigned short* __restrict__ Whh,  // (1024, 256) bf16
       const float* __restrict__ starts,        // (T, B)
       const float* __restrict__ h0,            // (B, HID)
       const float* __restrict__ c0,
       float* __restrict__ HOUT) {              // (TB, HID)
  __shared__ unsigned short h_bf[BB * HID];
  __shared__ float gates[BB * 1024];
  __shared__ float hst[BB * HID];
  __shared__ float cst[BB * HID];

  const int tid  = threadIdx.x;
  const int lane = tid & 31;
  const int wv   = tid >> 5;
  const int nl   = lane & 15;
  const int mhi  = (lane & 16) ? 8 : 0;

  for (int i = tid; i < BB * HID; i += 256) {
    float h = h0[i]; hst[i] = h; h_bf[i] = f2bf(h); cst[i] = c0[i];
  }
  __syncthreads();

  for (int t = 0; t < TT; ++t) {
    // episode-start masking
    for (int i = tid; i < BB * HID; i += 256) {
      float keep = 1.0f - starts[t * BB + (i >> 8)];
      float h = hst[i] * keep;
      hst[i] = h; h_bf[i] = f2bf(h); cst[i] *= keep;
    }
    __syncthreads();

    // gates += h @ Whh^T : M=16, N=1024 (wave wv covers n in [wv*128, wv*128+128))
    v8f acc[8];
#pragma unroll
    for (int q = 0; q < 8; ++q) acc[q] = v8f_zero();
    for (int kk = 0; kk < HID / 32; ++kk) {
      Frag16 a;
      const unsigned short* ap = h_bf + (lane & 15) * HID + kk * 32;
#pragma unroll
      for (int v = 0; v < 8; ++v)
        a.u[v] = *(const unsigned int*)(ap + a_kofs(lane, v));
#pragma unroll
      for (int q = 0; q < 8; ++q) {
        Frag16 b;
        const unsigned short* bp =
            Whh + (size_t)(wv * 128 + q * 16 + nl) * HID + kk * 32;
#pragma unroll
        for (int v = 0; v < 8; ++v)
          b.u[v] = *(const unsigned int*)(bp + b_kofs(lane, v));
        acc[q] = wmma_bf16(a, b, acc[q]);
      }
    }
#pragma unroll
    for (int q = 0; q < 8; ++q) {
      int n = wv * 128 + q * 16 + nl;
#pragma unroll
      for (int v = 0; v < 8; ++v) gates[(mhi + v) * 1024 + n] = acc[q][v];
    }
    __syncthreads();

    // activations + state update
    const float* gx = GX + (size_t)t * BB * 1024;
    for (int i = tid; i < BB * HID; i += 256) {
      int b = i >> 8; int j = i & 255;
      const float* gb  = gates + b * 1024;
      const float* gxb = gx + b * 1024;
      float gi = gb[j]        + gxb[j];
      float gf = gb[256 + j]  + gxb[256 + j];
      float gg = gb[512 + j]  + gxb[512 + j];
      float go = gb[768 + j]  + gxb[768 + j];
      float si = 1.0f / (1.0f + __expf(-gi));
      float sf = 1.0f / (1.0f + __expf(-gf));
      float so = 1.0f / (1.0f + __expf(-go));
      float c  = sf * cst[i] + si * tanhf(gg);
      float h  = so * tanhf(c);
      cst[i] = c; hst[i] = h; h_bf[i] = f2bf(h);
      HOUT[(size_t)t * (BB * HID) + i] = h;
    }
    __syncthreads();
  }
}

// ---------------------------------------------------------------------------
// Heads: actor log-softmax / entropy / critic / aux. One thread per (t,b) row.
// ---------------------------------------------------------------------------
__global__ void __launch_bounds__(256)
k_heads(const float* __restrict__ HOUT,              // (TB, HID)
        const unsigned short* __restrict__ FEATbf,   // (TB, FEAT) bf16
        const int* __restrict__ actions,             // (TB,)
        const float* __restrict__ actor_w, const float* __restrict__ actor_b,
        const float* __restrict__ critic_w, const float* __restrict__ critic_b,
        const float* __restrict__ aux_w, const float* __restrict__ aux_b,
        float* __restrict__ out) {
  int r = blockIdx.x * 256 + threadIdx.x;
  if (r >= TB) return;
  const float* h = HOUT + (size_t)r * HID;

  float logits[NUM_ACTIONS];
  float mx = -3.4e38f;
  for (int a = 0; a < NUM_ACTIONS; ++a) {
    const float* w = actor_w + a * HID;
    float s = actor_b[a];
    for (int k = 0; k < HID; ++k) s += h[k] * w[k];
    logits[a] = s;
    mx = s > mx ? s : mx;
  }
  float se = 0.0f;
  for (int a = 0; a < NUM_ACTIONS; ++a) se += __expf(logits[a] - mx);
  float lse = mx + __logf(se);

  float ent = 0.0f;
  for (int a = 0; a < NUM_ACTIONS; ++a) {
    float lp = logits[a] - lse;
    ent -= __expf(lp) * lp;
  }
  float val = critic_b[0];
  for (int k = 0; k < HID; ++k) val += h[k] * critic_w[k];

  float aux = aux_b[0];
  const unsigned short* f = FEATbf + (size_t)r * FEAT;
  for (int k = 0; k < FEAT; ++k) aux += bf2f(f[k]) * aux_w[k];

  int act = actions[r];
  out[r]            = logits[act] - lse;   // log_probs
  out[TB + r]       = ent;                 // entropy
  out[2 * TB + r]   = val;                 // values
  out[3 * TB + r]   = aux;                 // aux_logits
}

// ---------------------------------------------------------------------------
// Workspace layout (bytes, all 256-aligned)
// ---------------------------------------------------------------------------
#define OFF_X0    ((size_t)0)                       // obs bf16 NHWC (2048,84,84,4)
#define OFF_A1    (OFF_X0   + 115605504)            // conv1 out (2048,20,20,32) bf16
#define OFF_A2    (OFF_A1   + 52428800)             // conv2 out (2048,9,9,64)   bf16
#define OFF_A3    (OFF_A2   + 21233664)             // conv3 out (2048,7,7,64)   bf16
#define OFF_FEATB (OFF_A3   + 12845056)             // feat (2048,512) bf16
#define OFF_GX    (OFF_FEATB + 2097152)             // gates_x (2048,1024) f32
#define OFF_HOUT  (OFF_GX   + 8388608)              // h out (2048,256) f32
#define OFF_W1    (OFF_HOUT + 2097152)              // conv1 w bf16 (32,256)
#define OFF_W2    (OFF_W1   + 16384)                // conv2 w bf16 (64,512)
#define OFF_W3    (OFF_W2   + 65536)                // conv3 w bf16 (64,576)
#define OFF_WFC   (OFF_W3   + 73728)                // fc w bf16 (512,3136)
#define OFF_WIH   (OFF_WFC  + 3211264)              // w_ih bf16 (1024,512)
#define OFF_WHH   (OFF_WIH  + 1048576)              // w_hh bf16 (1024,256)

extern "C" void kernel_launch(void* const* d_in, const int* in_sizes, int n_in,
                              void* d_out, int out_size, void* d_ws, size_t ws_size,
                              hipStream_t stream) {
  const float* obs     = (const float*)d_in[0];
  const int*   actions = (const int*)d_in[1];
  const float* starts  = (const float*)d_in[2];
  const float* h0      = (const float*)d_in[3];
  const float* c0      = (const float*)d_in[4];
  const float* c1w = (const float*)d_in[5];  const float* c1b = (const float*)d_in[6];
  const float* c2w = (const float*)d_in[7];  const float* c2b = (const float*)d_in[8];
  const float* c3w = (const float*)d_in[9];  const float* c3b = (const float*)d_in[10];
  const float* fcw = (const float*)d_in[11]; const float* fcb = (const float*)d_in[12];
  const float* wih = (const float*)d_in[13]; const float* whh = (const float*)d_in[14];
  const float* bih = (const float*)d_in[15]; const float* bhh = (const float*)d_in[16];
  const float* aw  = (const float*)d_in[17]; const float* ab  = (const float*)d_in[18];
  const float* cw  = (const float*)d_in[19]; const float* cb  = (const float*)d_in[20];
  const float* xw  = (const float*)d_in[21]; const float* xb  = (const float*)d_in[22];

  char* ws = (char*)d_ws;
  unsigned short* X0    = (unsigned short*)(ws + OFF_X0);
  unsigned short* A1    = (unsigned short*)(ws + OFF_A1);
  unsigned short* A2    = (unsigned short*)(ws + OFF_A2);
  unsigned short* A3    = (unsigned short*)(ws + OFF_A3);
  unsigned short* FEATB = (unsigned short*)(ws + OFF_FEATB);
  float*          GX    = (float*)(ws + OFF_GX);
  float*          HOUT  = (float*)(ws + OFF_HOUT);
  unsigned short* W1    = (unsigned short*)(ws + OFF_W1);
  unsigned short* W2    = (unsigned short*)(ws + OFF_W2);
  unsigned short* W3    = (unsigned short*)(ws + OFF_W3);
  unsigned short* WFC   = (unsigned short*)(ws + OFF_WFC);
  unsigned short* WIH   = (unsigned short*)(ws + OFF_WIH);
  unsigned short* WHH   = (unsigned short*)(ws + OFF_WHH);

  // --- weight prep (independent) ---
  k_prep_conv_w<32, 4, 8, 8><<<32, 256, 0, stream>>>(c1w, W1);
  k_prep_conv_w<64, 32, 4, 4><<<128, 256, 0, stream>>>(c2w, W2);
  k_prep_conv_w<64, 64, 3, 3><<<144, 256, 0, stream>>>(c3w, W3);
  k_prep_fc_w<<<6272, 256, 0, stream>>>(fcw, WFC);
  k_prep_plain<<<2048, 256, 0, stream>>>(wih, WIH, 1024 * 512);
  k_prep_plain<<<1024, 256, 0, stream>>>(whh, WHH, 1024 * 256);

  // --- obs -> bf16 NHWC ---
  k_obs_to_bf16<<<225792, 256, 0, stream>>>(obs, X0);

  // --- conv stack (implicit GEMM, WMMA, async weight staging) ---
  k_conv_wmma<4, 84, 84, 8, 8, 4, 20, 20, 32><<<6400, 256, 0, stream>>>(X0, W1, c1b, A1);
  k_conv_wmma<32, 20, 20, 4, 4, 2, 9, 9, 64><<<1296, 256, 0, stream>>>(A1, W2, c2b, A2);
  k_conv_wmma<64, 9, 9, 3, 3, 1, 7, 7, 64><<<784, 256, 0, stream>>>(A2, W3, c3b, A3);

  // --- FC: feat = relu(conv3flat @ fc_w^T + b), bf16 out ---
  k_gemm_bf16<3136, FEAT, true, false><<<16, 256, 0, stream>>>(
      A3, WFC, fcb, nullptr, FEATB, nullptr);

  // --- LSTM input-side GEMM: GX = feat @ w_ih^T + b_ih + b_hh (f32) ---
  k_gemm_bf16<FEAT, 1024, false, true><<<16, 256, 0, stream>>>(
      FEATB, WIH, bih, bhh, nullptr, GX);

  // --- recurrent scan (persistent single block) ---
  k_lstm<<<1, 256, 0, stream>>>(GX, WHH, starts, h0, c0, HOUT);

  // --- heads (+aux) ---
  k_heads<<<8, 256, 0, stream>>>(HOUT, FEATB, actions, aw, ab, cw, cb, xw, xb,
                                 (float*)d_out);
}
